// SACRNN_83992380441068
// MI455X (gfx1250) — matrix operand order repaired and checked by reading, not compiled, not run on hardware
//
#include <hip/hip_runtime.h>
#include <math.h>

// ---------------- types ----------------
typedef __attribute__((ext_vector_type(16))) __bf16 v16bf;
typedef __attribute__((ext_vector_type(8)))  __bf16 v8bf;
typedef __attribute__((ext_vector_type(8)))  float  v8f;

#define DEVFN static __device__ __forceinline__

// ---------------- problem constants ----------------
constexpr int B  = 256;
constexpr int T  = 512;
constexpr int F  = 76;
constexpr int H  = 256;
constexpr int KW = 10;                 // window length K
constexpr int NZ  = 4*H + 2*4;         // 1032
constexpr int NZP = 1040;              // padded N for GEMM1 (65 tiles of 16)
constexpr int K1P = 352;               // padded K for GEMM1 (>=334, 11 chunks of 32)
constexpr int K2  = KW*H;              // 2560 (80 chunks of 32)
constexpr int NT1 = 65;                // n-tiles in GEMM1

// ---------------- workspace layout (bytes) ----------------
constexpr size_t WKR_OFF   = 0;                                  // ushort [NZP][K1P], col-major
constexpr size_t WKR_SZ    = (size_t)NZP*K1P*2;
constexpr size_t WCT_OFF   = WKR_OFF + WKR_SZ;                   // ushort [H][K2], col-major (per o)
constexpr size_t WCT_SZ    = (size_t)H*K2*2;
constexpr size_t A1_OFF    = WCT_OFF + WCT_SZ;                   // ushort [B][K1P]
constexpr size_t A1_SZ     = (size_t)B*K1P*2;
constexpr size_t BIASZ_OFF = A1_OFF + A1_SZ;                     // float [NZ]
constexpr size_t BIASZ_SZ  = ((NZ*4 + 255)/256)*256;
constexpr size_t CST_OFF   = BIASZ_OFF + BIASZ_SZ;               // float c[B][H]
constexpr size_t WINH_OFF  = CST_OFF + (size_t)B*H*4;            // float [KW][B][H]
constexpr size_t WIND_OFF  = WINH_OFF + (size_t)KW*B*H*4;        // float [KW][B]
constexpr size_t WS_END    = WIND_OFF + (size_t)KW*B*4;
constexpr size_t ZERO_CNT  = (WS_END - CST_OFF)/4;               // floats zeroed each call

// ---------------- LDS layout (bytes, dynamic) ----------------
constexpr size_t LDS_A2   = 0;                        // bf16 16 x 2560 = 81920 (first 11264 double as A1 stage)
constexpr size_t LDS_Z    = 81920;                    // f32 16 x 1032 = 66048
constexpr size_t LDS_H    = LDS_Z + 66048;            // f32 16 x 256  = 16384
constexpr size_t LDS_CURD = LDS_H + 16384;            // f32 16        = 64
constexpr size_t LDS_SRED = LDS_CURD + 64;            // f32 16 x 16   = 1024
constexpr size_t LDS_TOTAL= LDS_SRED + 1024;          // ~165.4 KB

// ---------------- helpers ----------------
DEVFN unsigned short f2bf(float f) {
  union { float f; unsigned u; } v; v.f = f;
  unsigned u = v.u;
  return (unsigned short)((u + 0x7FFFu + ((u >> 16) & 1u)) >> 16); // RNE
}
DEVFN v16bf mk16(v8bf lo, v8bf hi) {
  return __builtin_shufflevector(lo, hi, 0,1,2,3,4,5,6,7,8,9,10,11,12,13,14,15);
}
DEVFN float sigm(float x) { return 1.0f / (1.0f + __expf(-x)); }

// ================= one-time prep kernels =================

// Wkr: rows [Wk(0..75); Wk[76]; Wr(0..255); Wr[256]; zero pad], stored col-major
__global__ void k_build_wkr(const float* __restrict__ Wk, const float* __restrict__ Wr,
                            const float* __restrict__ bk, const float* __restrict__ br,
                            unsigned short* __restrict__ wkr, float* __restrict__ biasz) {
  int j = blockIdx.x;            // 0..NZP-1
  int r = threadIdx.x;           // 0..K1P-1
  float v = 0.f;
  if (j < NZ) {
    if (r < 76)        v = Wk[(size_t)r*NZ + j];
    else if (r == 76)  v = Wk[(size_t)76*NZ + j];
    else if (r < 333)  v = Wr[(size_t)(r-77)*NZ + j];
    else if (r == 333) v = Wr[(size_t)256*NZ + j];
    if (r == 0) biasz[j] = bk[j] + br[j];
  }
  wkr[(size_t)j*K1P + r] = f2bf(v);
}

// WcT[o][k*H + c] = Wc[o][c][k]  (Wc is (H,H,KW) row-major)
__global__ void k_build_wct(const float* __restrict__ Wc, unsigned short* __restrict__ wct) {
  int o = blockIdx.x, c = threadIdx.x;
  #pragma unroll
  for (int k = 0; k < KW; ++k)
    wct[(size_t)o*K2 + k*H + c] = f2bf(Wc[((size_t)o*H + c)*KW + k]);
}

__global__ void k_zero(float* __restrict__ p, size_t n) {
  size_t i = (size_t)blockIdx.x*blockDim.x + threadIdx.x;
  if (i < n) p[i] = 0.f;
}

// A1 for t=0: [x(:,0,:), time(:,0), h=0, time(:,0), pad 0]
__global__ void k_init_a1(const float* __restrict__ x, const float* __restrict__ timev,
                          unsigned short* __restrict__ a1) {
  int b = blockIdx.x, idx = threadIdx.x;
  float v = 0.f;
  if (idx < 76)                      v = x[((size_t)b*T + 0)*F + idx];
  else if (idx == 76 || idx == 333)  v = timev[(size_t)b*T + 0];
  a1[(size_t)b*K1P + idx] = f2bf(v);
}

// ================= fused per-step kernel =================
// grid = 16 (m-tiles of 16 batches), block = 512 (16 waves)
// phases: GEMM1(z) -> gates/cell -> window softmax + A2(LDS) -> GEMM2(conv) -> fused output
__global__ void __launch_bounds__(512)
k_step(unsigned short* __restrict__ a1,        // [B][K1P] read this step, rewritten for t+1
       const unsigned short* __restrict__ wkr,
       const unsigned short* __restrict__ wct,
       const float* __restrict__ biasz,
       const float* __restrict__ x, const float* __restrict__ timev,
       float* __restrict__ cws,
       float* __restrict__ winh, float* __restrict__ wind,
       const float* __restrict__ bc, const float* __restrict__ Wo,
       const float* __restrict__ bo,
       float* __restrict__ outp, float* __restrict__ distp, int t) {
  extern __shared__ __align__(16) char smem[];
  unsigned short* a2s  = (unsigned short*)(smem + LDS_A2);   // 16 x K2 bf16
  unsigned short* a1s  = (unsigned short*)(smem + LDS_A2);   // overlay: A1 stage (dead after hoist)
  float*          zl   = (float*)(smem + LDS_Z);             // 16 x NZ f32
  float*          hls  = (float*)(smem + LDS_H);             // 16 x H f32 (h_new tile)
  float*          curd = (float*)(smem + LDS_CURD);
  float*          sred = (float*)(smem + LDS_SRED);

  const int tid = threadIdx.x;
  const int wg  = blockIdx.x;        // m-tile
  const int wv  = tid >> 5;          // wave id 0..15
  const int ln  = tid & 31;
  const int l16 = ln & 15;
  const int grp = ln >> 4;

  // ---- phase 0: stage A1 tile into LDS (16 rows x K1P bf16) ----
  {
    const uint4* src = (const uint4*)(a1 + (size_t)wg*16*K1P);
    uint4* dst = (uint4*)a1s;
    for (int i = tid; i < 16*K1P/8; i += 512) dst[i] = src[i];
  }
  __syncthreads();

  // ---- hoist A fragments (11 k-chunks); a1s is dead afterwards ----
  v16bf afr[11];
  #pragma unroll
  for (int kc = 0; kc < 11; ++kc) {
    const unsigned short* p = a1s + (size_t)l16*K1P + kc*32 + grp*8;
    afr[kc] = mk16(*(const v8bf*)p, *(const v8bf*)(p + 16));
  }

  // ---- phase 1: GEMM1, wave owns n-tiles wv, wv+16, ... (65 tiles) ----
  for (int nt = wv; nt < NT1; nt += 16) {
    v8f acc = {0.f,0.f,0.f,0.f,0.f,0.f,0.f,0.f};
    const unsigned short* bcol = wkr + (size_t)(nt*16 + l16)*K1P + grp*8;
    #pragma unroll
    for (int kc = 0; kc < 11; ++kc) {
      const unsigned short* p = bcol + kc*32;
      v16bf bfr = mk16(*(const v8bf*)p, *(const v8bf*)(p + 16));
      acc = __builtin_amdgcn_wmma_f32_16x16x32_bf16(false, afr[kc], false, bfr,
                                                    (short)0, acc, false, false);
    }
    int j = nt*16 + l16;               // z column
    if (j < NZ) {
      float bz = biasz[j];
      #pragma unroll
      for (int r = 0; r < 8; ++r) {
        int M = r + 8*grp;
        zl[(size_t)M*NZ + j] = acc[r] + bz;
      }
    }
  }
  __syncthreads();

  // ---- phase 2: gates / cell update, one wave per batch row ----
  const int bl = wv;
  const int b  = wg*16 + bl;
  const float* zr = zl + (size_t)bl*NZ;

  // fm = cumsum(softmax(z[0:4]))
  float z0 = zr[0], z1 = zr[1], z2 = zr[2], z3 = zr[3];
  float m1 = fmaxf(fmaxf(z0,z1), fmaxf(z2,z3));
  float e0 = __expf(z0-m1), e1 = __expf(z1-m1), e2 = __expf(z2-m1), e3 = __expf(z3-m1);
  float inv1 = 1.f/(e0+e1+e2+e3);
  float fmv[4];
  fmv[0] = e0*inv1; fmv[1] = fmv[0] + e1*inv1; fmv[2] = fmv[1] + e2*inv1; fmv[3] = fmv[2] + e3*inv1;
  // im[i] = sum_{m>=i} softmax(z[4:8])[m]
  float y0 = zr[4], y1 = zr[5], y2 = zr[6], y3 = zr[7];
  float m2 = fmaxf(fmaxf(y0,y1), fmaxf(y2,y3));
  float q0 = __expf(y0-m2), q1 = __expf(y1-m2), q2 = __expf(y2-m2), q3 = __expf(y3-m2);
  float inv2 = 1.f/(q0+q1+q2+q3);
  float imv[4];
  imv[3] = q3*inv2; imv[2] = imv[3] + q2*inv2; imv[1] = imv[2] + q1*inv2; imv[0] = imv[1] + q0*inv2;
  float cur_dist = 1.f - 0.25f*(fmv[0]+fmv[1]+fmv[2]+fmv[3]);

  const int slot = t % KW;
  float hreg[8];
  #pragma unroll
  for (int ii = 0; ii < 8; ++ii) {
    int c = ln + 32*ii;                // hidden index; c == l*64+m
    int l = c >> 6;
    float fg = sigm(zr[8 + c]);
    float ig = sigm(zr[8 + 256 + c]);
    float og = sigm(zr[8 + 512 + c]);
    float ci = tanhf(zr[8 + 768 + c]);
    float cl = cws[(size_t)b*H + c];
    float fv = fmv[l], iv = imv[l];
    float ov = fv*iv;
    float cn = ov*(fg*cl + ig*ci) + (fv-ov)*cl + (iv-ov)*ci;
    cws[(size_t)b*H + c] = cn;
    float hn = og * tanhf(cn);
    hreg[ii] = hn;
    hls[(size_t)bl*H + c] = hn;                         // LDS h tile for epilogue
    winh[(size_t)slot*B*H + (size_t)b*H + c] = hn;      // cross-step window state
    if (t+1 < T) a1[(size_t)b*K1P + 77 + c] = f2bf(hn); // next-step A1 (h section)
  }
  if (ln == 0) {
    wind[(size_t)slot*B + b] = cur_dist;
    curd[bl] = cur_dist;
    distp[(size_t)t*B + b] = cur_dist;
  }
  if (t+1 < T) {
    for (int idx = ln; idx < 77; idx += 32) {
      float v = (idx < 76) ? x[((size_t)b*T + (t+1))*F + idx] : timev[(size_t)b*T + (t+1)];
      a1[(size_t)b*K1P + idx] = f2bf(v);
    }
    if (ln == 0) a1[(size_t)b*K1P + 333] = f2bf(timev[(size_t)b*T + (t+1)]);
  }
  __syncthreads();

  // ---- phase 3: window softmax ld, build A2 tile directly in LDS ----
  float d[KW];
  #pragma unroll
  for (int j = 0; j < KW-1; ++j) d[j] = wind[(size_t)((t+1+j)%KW)*B + b];
  d[KW-1] = curd[bl];
  float a[KW], cs = 0.f;
  #pragma unroll
  for (int j = 0; j < KW; ++j) { cs += d[j]; a[j] = cs; }
  float mx = a[0];
  #pragma unroll
  for (int j = 1; j < KW; ++j) mx = fmaxf(mx, a[j]);
  float ldw[KW], se = 0.f;
  #pragma unroll
  for (int k = 0; k < KW; ++k) { float e = __expf(a[KW-1-k] - mx); ldw[k] = e; se += e; }
  float invs = 1.f/se;

  #pragma unroll
  for (int j = 0; j < KW; ++j) {
    float wgt = ldw[j]*invs;
    size_t base = (size_t)bl*K2 + (size_t)j*H;
    if (j == KW-1) {
      #pragma unroll
      for (int ii = 0; ii < 8; ++ii) a2s[base + ln + 32*ii] = f2bf(hreg[ii]*wgt);
    } else {
      const float* hp = winh + (size_t)((t+1+j)%KW)*B*H + (size_t)b*H;
      #pragma unroll
      for (int ii = 0; ii < 8; ++ii) { int c = ln + 32*ii; a2s[base + c] = f2bf(hp[c]*wgt); }
    }
  }
  __syncthreads();

  // ---- phase 4: GEMM2 (conv), wave wv owns n-tile wv ----
  v8f acc = {0.f,0.f,0.f,0.f,0.f,0.f,0.f,0.f};
  const unsigned short* arow = a2s + (size_t)l16*K2;
  const unsigned short* bcol = wct + (size_t)(wv*16 + l16)*K2 + grp*8;
  #pragma unroll 4
  for (int kc = 0; kc < K2/32; ++kc) {
    const unsigned short* ap = arow + kc*32 + grp*8;
    v16bf af = mk16(*(const v8bf*)ap, *(const v8bf*)(ap + 16));
    const unsigned short* bp = bcol + kc*32;
    v16bf bf = mk16(*(const v8bf*)bp, *(const v8bf*)(bp + 16));
    if (kc + 4 < K2/32) __builtin_prefetch(bp + 4*32, 0, 1);    // global_prefetch_b8
    acc = __builtin_amdgcn_wmma_f32_16x16x32_bf16(false, af, false, bf,
                                                  (short)0, acc, false, false);
  }

  // ---- phase 5: epilogue: conv+bc+h, dot with Wo, sigmoid ----
  const int o  = wv*16 + l16;
  const float wo  = Wo[o];
  const float bcv = bc[o];
  float p[8];
  #pragma unroll
  for (int r = 0; r < 8; ++r) {
    int M = r + 8*grp;
    float rnn = acc[r] + bcv + hls[(size_t)M*H + o];
    p[r] = rnn * wo;
  }
  #pragma unroll
  for (int r = 0; r < 8; ++r) {
    float v = p[r];
    for (int m = 1; m < 16; m <<= 1) v += __shfl_xor(v, m, 32);  // fixed-order, deterministic
    p[r] = v;
  }
  if (l16 == 0) {
    #pragma unroll
    for (int r = 0; r < 8; ++r) sred[(r + 8*grp)*16 + wv] = p[r];
  }
  __syncthreads();
  if (tid < 16) {
    float s = 0.f;
    #pragma unroll
    for (int w = 0; w < 16; ++w) s += sred[tid*16 + w];
    int bb = wg*16 + tid;
    outp[(size_t)bb*T + t] = 1.f/(1.f + __expf(-(s + bo[0])));
  }
}

// ================= host =================
extern "C" void kernel_launch(void* const* d_in, const int* in_sizes, int n_in,
                              void* d_out, int out_size, void* d_ws, size_t ws_size,
                              hipStream_t stream) {
  (void)in_sizes; (void)n_in; (void)out_size; (void)ws_size;
  const float* x     = (const float*)d_in[0];
  const float* timev = (const float*)d_in[1];
  const float* Wk    = (const float*)d_in[2];
  const float* bk    = (const float*)d_in[3];
  const float* Wr    = (const float*)d_in[4];
  const float* br    = (const float*)d_in[5];
  const float* Wc    = (const float*)d_in[6];
  const float* bc    = (const float*)d_in[7];
  const float* Wo    = (const float*)d_in[8];
  const float* bo    = (const float*)d_in[9];

  char* ws = (char*)d_ws;
  unsigned short* wkr   = (unsigned short*)(ws + WKR_OFF);
  unsigned short* wct   = (unsigned short*)(ws + WCT_OFF);
  unsigned short* a1    = (unsigned short*)(ws + A1_OFF);
  float*          biasz = (float*)(ws + BIASZ_OFF);
  float*          cws   = (float*)(ws + CST_OFF);
  float*          winh  = (float*)(ws + WINH_OFF);
  float*          wind  = (float*)(ws + WIND_OFF);

  float* outp  = (float*)d_out;
  float* distp = outp + (size_t)B*T;   // output tuple: out (B,T,1) then dist (T,B)

  // one-time prep (re-run every call -> deterministic)
  k_build_wkr<<<NZP, K1P, 0, stream>>>(Wk, Wr, bk, br, wkr, biasz);
  k_build_wct<<<H, H, 0, stream>>>(Wc, wct);
  k_zero<<<(int)((ZERO_CNT + 255)/256), 256, 0, stream>>>(cws, ZERO_CNT);
  k_init_a1<<<B, K1P, 0, stream>>>(x, timev, a1);

  for (int t = 0; t < T; ++t) {
    k_step<<<16, 512, LDS_TOTAL, stream>>>(a1, wkr, wct, biasz, x, timev,
                                           cws, winh, wind, bc, Wo, bo,
                                           outp, distp, t);
  }
}
